// FusionAttention2_1606317769294
// MI455X (gfx1250) — matrix operand order repaired and checked by reading, not compiled, not run on hardware
//
#include <hip/hip_runtime.h>

#define NN 4096
#define DD 512

typedef __attribute__((ext_vector_type(16))) _Float16 v16h;
typedef __attribute__((ext_vector_type(8)))  _Float16 v8h;
typedef __attribute__((ext_vector_type(8)))  float    v8f;
typedef __attribute__((ext_vector_type(4)))  unsigned int u32x4;
typedef __attribute__((ext_vector_type(8)))  int      i32x8;
typedef __attribute__((ext_vector_type(4)))  int      i32x4;

union F16Frag { v16h v; v8h h[2]; };

static __device__ inline v8f zero_v8f() {
    v8f z;
#pragma unroll
    for (int i = 0; i < 8; ++i) z[i] = 0.0f;
    return z;
}

static __device__ inline v8f wmma_f16(F16Frag a, v16h b, v8f c) {
    // D = A(16x32 f16) x B(32x16 f16) + C(16x16 f32)
    return __builtin_amdgcn_wmma_f32_16x16x32_f16(
        /*neg_a=*/false, a.v, /*neg_b=*/false, b,
        /*c_mod=*/(short)0, c, /*reuse_a=*/false, /*reuse_b=*/false);
}

// ---------------------------------------------------------------- TDM support
#if defined(__has_builtin)
#if __has_builtin(__builtin_amdgcn_tensor_load_to_lds) && \
    __has_builtin(__builtin_amdgcn_s_wait_tensorcnt)
#define HAVE_TDM 1
#endif
#endif

#ifdef HAVE_TDM
typedef __attribute__((address_space(3))) _Float16 lds_f16;

static __device__ inline unsigned lds_byte_off(const _Float16* p) {
    // generic -> LDS address space -> 32-bit LDS byte offset
    return (unsigned)(unsigned long long)(lds_f16*)p;
}

// Issue a 2D TDM tile load: tile_x elements (2B each) x tile_y rows,
// source rows strided by stride_x elements, into LDS at lds_off (compact).
static __device__ inline void tdm_issue_2d(const _Float16* gaddr, unsigned lds_off,
                                           int tile_x, int tile_y,
                                           int tensor_x, int tensor_y, int stride_x) {
    unsigned long long ga = (unsigned long long)gaddr;
    u32x4 g0;
    g0.x = 1u;                                   // count=1 (valid), user mode
    g0.y = lds_off;                              // lds_addr (bytes)
    g0.z = (unsigned)(ga & 0xffffffffu);         // global_addr[31:0]
    g0.w = (unsigned)((ga >> 32) & 0x1ffffffu)   // global_addr[56:32]
         | (2u << 30);                           // type = 2 ("image")
    i32x8 g1;
    g1[0] = (1 << 16);                           // data_size = 1 -> 2 bytes
    g1[1] = (tensor_x & 0xffff) << 16;           // tensor_dim0[15:0]
    g1[2] = ((tensor_x >> 16) & 0xffff) | ((tensor_y & 0xffff) << 16);
    g1[3] = ((tensor_y >> 16) & 0xffff) | ((tile_x & 0xffff) << 16);
    g1[4] = (tile_y & 0xffff);                   // tile_dim1; tile_dim2 = 0
    g1[5] = stride_x;                            // tensor_dim0_stride[31:0]
    g1[6] = 0;
    g1[7] = 0;
    i32x4 z4 = {0, 0, 0, 0};
#if __clang_major__ >= 23
    i32x8 z8 = {0, 0, 0, 0, 0, 0, 0, 0};
    __builtin_amdgcn_tensor_load_to_lds(g0, g1, z4, z4, z8, 0);
#else
    __builtin_amdgcn_tensor_load_to_lds(g0, g1, z4, z4, 0);
#endif
}
#endif  // HAVE_TDM

// ---------------------------------------------------------------- converts
__global__ __launch_bounds__(256) void f32_to_f16_kernel(
    const float* __restrict__ in, _Float16* __restrict__ out, int n) {
    int i = blockIdx.x * 256 + threadIdx.x;
    if (i < n) out[i] = (_Float16)in[i];
}

// out[j*d + k] = (f16) in[k*d + j]   (W -> W^T, f32 -> f16)
__global__ __launch_bounds__(1024) void transpose_w_kernel(
    const float* __restrict__ in, _Float16* __restrict__ out, int d) {
    __shared__ float tile[32][33];
    int x = blockIdx.x * 32 + threadIdx.x;
    int y = blockIdx.y * 32 + threadIdx.y;
    tile[threadIdx.y][threadIdx.x] = in[(size_t)y * d + x];
    __syncthreads();
    int ox = blockIdx.y * 32 + threadIdx.x;
    int oy = blockIdx.x * 32 + threadIdx.y;
    out[(size_t)oy * d + ox] = (_Float16)tile[threadIdx.x][threadIdx.y];
}

// ---------------------------------------------------------------- GEMM
// C[M,Nc] = act(A[M,K] @ B + bias), B given transposed: BT[Nc,K] row-major.
// 4 waves per block; wave w computes rows [by*64 + 16w, +16) x cols [bx*64, +64).
// The shared 64x32 B tile is staged into LDS by the Tensor Data Mover
// (double-buffered, one wave issues, TENSORcnt + barrier sync); A fragments
// stream from global per wave.
__global__ __launch_bounds__(128) void gemm_wmma_kernel(
    const _Float16* __restrict__ A, const _Float16* __restrict__ BT,
    const float* __restrict__ bias,
    _Float16* __restrict__ outH,   // f16 [M,Nc] row-major (or null)
    _Float16* __restrict__ outT,   // f16 [Nc,M] (transposed) (or null)
    float* __restrict__ outF,      // f32 [M,Nc] row-major (or null)
    int M, int K, int Nc, int relu) {
    const int lane = threadIdx.x & 31;
    const int wv   = threadIdx.x >> 5;
    const int r    = lane & 15;
    const int hi   = lane >> 4;
    const int row0 = blockIdx.y * 64 + wv * 16;
    const int col0 = blockIdx.x * 64;

    v8f acc[4];
#pragma unroll
    for (int t = 0; t < 4; ++t) acc[t] = zero_v8f();

    const _Float16* Ap = A + (size_t)(row0 + r) * K + hi * 16;

#ifdef HAVE_TDM
    // [buf][row 0..63][halves 0..31] compact tile
    __shared__ __align__(64) _Float16 sB[2][64 * 32];
    const int NT = K / 32;
    if (wv == 0)
        tdm_issue_2d(BT + (size_t)col0 * K, lds_byte_off(&sB[0][0]),
                     32, 64, K, Nc, K);
    for (int kt = 0; kt < NT; ++kt) {
        const int k = kt * 32;
        if (wv == 0) __builtin_amdgcn_s_wait_tensorcnt(0);
        __syncthreads();  // tile kt ready; prior reads of buf[(kt+1)&1] done
        if (wv == 0 && kt + 1 < NT)
            tdm_issue_2d(BT + (size_t)col0 * K + (kt + 1) * 32,
                         lds_byte_off(&sB[(kt + 1) & 1][0]),
                         32, 64, K, Nc, K);
        F16Frag af;
        af.h[0] = *(const v8h*)(Ap + k);
        af.h[1] = *(const v8h*)(Ap + k + 8);
        const _Float16* bt = &sB[kt & 1][0];
#pragma unroll
        for (int t = 0; t < 4; ++t) {
            v16h bf = *(const v16h*)(bt + (t * 16 + r) * 32 + hi * 16);
            acc[t] = wmma_f16(af, bf, acc[t]);
        }
    }
#else
    for (int k = 0; k < K; k += 32) {
        F16Frag af;
        af.h[0] = *(const v8h*)(Ap + k);
        af.h[1] = *(const v8h*)(Ap + k + 8);
#pragma unroll
        for (int t = 0; t < 4; ++t) {
            const _Float16* Bp = BT + (size_t)(col0 + t * 16 + r) * K + k + hi * 16;
            v16h bf = *(const v16h*)Bp;
            acc[t] = wmma_f16(af, bf, acc[t]);
        }
    }
#endif

#pragma unroll
    for (int t = 0; t < 4; ++t) {
        int col = col0 + t * 16 + r;
        float bv = bias ? bias[col] : 0.0f;
        float vals[8];
#pragma unroll
        for (int v = 0; v < 8; ++v) {
            float x = acc[t][v] + bv;
            if (relu) x = x > 0.0f ? x : 0.0f;
            vals[v] = x;
        }
        if (outH) {
#pragma unroll
            for (int v = 0; v < 8; ++v)
                outH[(size_t)(row0 + v + 8 * hi) * Nc + col] = (_Float16)vals[v];
        }
        if (outF) {
#pragma unroll
            for (int v = 0; v < 8; ++v)
                outF[(size_t)(row0 + v + 8 * hi) * Nc + col] = vals[v];
        }
        if (outT) {
            v8h pk;
#pragma unroll
            for (int v = 0; v < 8; ++v) pk[v] = (_Float16)vals[v];
            *(v8h*)(outT + (size_t)col * M + row0 + 8 * hi) = pk;
        }
    }
}

// ---------------------------------------------------------------- flash attention
// Out[N,D] = softmax_row(Q@K^T * scale + bias) @ V,  V given as VT[D,N].
// One block (4 waves) per 16 query rows. Wave w owns K-dim slice [128w,128w+128)
// for the score GEMM and output columns [128w, 128w+128) for the PV GEMM.
__global__ __launch_bounds__(128) void flash_attn_kernel(
    const _Float16* __restrict__ Q, const _Float16* __restrict__ Kh,
    const _Float16* __restrict__ VT, const float* __restrict__ Bias,
    float* __restrict__ Out, int gauss) {
    __shared__ float    sPart[4][16][32];
    __shared__ __align__(16) _Float16 sP[16][32];
    __shared__ float    sScale[16];
    __shared__ float    sL[16];

    const int lane = threadIdx.x & 31;
    const int wv   = threadIdx.x >> 5;
    const int r    = lane & 15;
    const int hi   = lane >> 4;
    const int i0   = blockIdx.x * 16;    // query rows
    const int ks   = wv * 128;           // this wave's K-dim / V-col slice
    const float sscale = 0.044194173824159216f;  // 1/sqrt(512)

    // Preload Q fragments for the 4 k-steps of this wave's slice.
    F16Frag qf[4];
    {
        const _Float16* Qp = Q + (size_t)(i0 + r) * DD + ks + hi * 16;
#pragma unroll
        for (int s = 0; s < 4; ++s) {
            qf[s].h[0] = *(const v8h*)(Qp + s * 32);
            qf[s].h[1] = *(const v8h*)(Qp + s * 32 + 8);
        }
    }

    v8f oacc[8];
#pragma unroll
    for (int t = 0; t < 8; ++t) oacc[t] = zero_v8f();

    float m_run = -1e30f;  // valid only in wave 0, lanes < 16
    float l_run = 0.0f;

    for (int j = 0; j < NN; j += 32) {
        // prefetch next K tile rows for this wave's slice (streams from L2/HBM)
        if (j + 32 < NN) {
            __builtin_prefetch(Kh + (size_t)(j + 32 + r + hi * 16) * DD + ks, 0, 1);
        }
        // ---- partial scores S16x32 over this wave's K slice
        v8f s0 = zero_v8f(), s1 = zero_v8f();
#pragma unroll
        for (int s = 0; s < 4; ++s) {
            int kk = ks + s * 32;
            F16Frag af = qf[s];
            v16h b0 = *(const v16h*)(Kh + (size_t)(j + r) * DD + kk + hi * 16);
            v16h b1 = *(const v16h*)(Kh + (size_t)(j + 16 + r) * DD + kk + hi * 16);
            s0 = wmma_f16(af, b0, s0);
            s1 = wmma_f16(af, b1, s1);
        }
#pragma unroll
        for (int v = 0; v < 8; ++v) {
            sPart[wv][v + 8 * hi][r]      = s0[v];
            sPart[wv][v + 8 * hi][16 + r] = s1[v];
        }
        __syncthreads();

        // ---- wave 0: cross-wave reduce + streaming softmax (lane = row)
        if (wv == 0 && lane < 16) {
            int row = lane;
            const float* bp = Bias + (size_t)(i0 + row) * NN + j;
            if (j + 32 < NN) __builtin_prefetch(bp + 32, 0, 1);  // bias is HBM-streamed
            float sc[32];
            float mx = m_run;
#pragma unroll
            for (int c = 0; c < 32; ++c) {
                float sv = sPart[0][row][c] + sPart[1][row][c] +
                           sPart[2][row][c] + sPart[3][row][c];
                float b = bp[c];
                if (gauss) b = __expf(-b * b * 0.005f);  // exp(-De^2 / (2*10^2))
                float x = sv * sscale + b;
                sc[c] = x;
                mx = fmaxf(mx, x);
            }
            float corr = __expf(m_run - mx);
            float sum = 0.0f;
#pragma unroll
            for (int c = 0; c < 32; ++c) {
                float pv = __expf(sc[c] - mx);
                sum += pv;
                sP[row][c] = (_Float16)pv;
            }
            l_run = l_run * corr + sum;
            m_run = mx;
            sScale[row] = corr;
            sL[row] = l_run;
        }
        __syncthreads();

        // ---- all waves: rescale accumulators, P(16x32) @ V-slice
        F16Frag pf;
        pf.h[0] = *(const v8h*)(&sP[r][hi * 16]);
        pf.h[1] = *(const v8h*)(&sP[r][hi * 16 + 8]);
        float rsc[8];
#pragma unroll
        for (int v = 0; v < 8; ++v) rsc[v] = sScale[v + 8 * hi];
#pragma unroll
        for (int t = 0; t < 8; ++t) {
#pragma unroll
            for (int v = 0; v < 8; ++v) oacc[t][v] *= rsc[v];
            int n0 = ks + t * 16;
            v16h bv = *(const v16h*)(VT + (size_t)(n0 + r) * NN + j + hi * 16);
            oacc[t] = wmma_f16(pf, bv, oacc[t]);
        }
        __syncthreads();  // protect sPart/sP for next iteration
    }

    float linv[8];
#pragma unroll
    for (int v = 0; v < 8; ++v) linv[v] = 1.0f / sL[v + 8 * hi];
#pragma unroll
    for (int t = 0; t < 8; ++t) {
        int col = ks + t * 16 + r;
#pragma unroll
        for (int v = 0; v < 8; ++v) {
            int row = i0 + v + 8 * hi;
            Out[(size_t)row * DD + col] = oacc[t][v] * linv[v];
        }
    }
}

// ---------------------------------------------------------------- residual + LN
// y = LN(A_row + B_row) * g + beta; writes f32 and optionally f16.
__global__ __launch_bounds__(256) void resid_ln_kernel(
    const float* __restrict__ A, const float* __restrict__ B,
    const float* __restrict__ g, const float* __restrict__ beta,
    float* __restrict__ outF, _Float16* __restrict__ outH) {
    __shared__ float redM[8];
    __shared__ float redV[8];
    const int row = blockIdx.x;
    const int t = threadIdx.x;
    const float* a = A + (size_t)row * DD;
    const float* b = B + (size_t)row * DD;
    float x0 = a[t] + b[t];
    float x1 = a[t + 256] + b[t + 256];

    float s = x0 + x1;
#pragma unroll
    for (int off = 16; off > 0; off >>= 1) s += __shfl_down(s, off, 32);
    if ((t & 31) == 0) redM[t >> 5] = s;
    __syncthreads();
    float mean = 0.0f;
#pragma unroll
    for (int i = 0; i < 8; ++i) mean += redM[i];
    mean *= (1.0f / DD);

    float d0 = x0 - mean, d1 = x1 - mean;
    float vs = d0 * d0 + d1 * d1;
#pragma unroll
    for (int off = 16; off > 0; off >>= 1) vs += __shfl_down(vs, off, 32);
    if ((t & 31) == 0) redV[t >> 5] = vs;
    __syncthreads();
    float var = 0.0f;
#pragma unroll
    for (int i = 0; i < 8; ++i) var += redV[i];
    var *= (1.0f / DD);

    float rstd = rsqrtf(var + 1e-5f);
    float y0 = d0 * rstd * g[t] + beta[t];
    float y1 = d1 * rstd * g[t + 256] + beta[t + 256];
    if (outF) {
        outF[(size_t)row * DD + t] = y0;
        outF[(size_t)row * DD + t + 256] = y1;
    }
    if (outH) {
        outH[(size_t)row * DD + t] = (_Float16)y0;
        outH[(size_t)row * DD + t + 256] = (_Float16)y1;
    }
}

__global__ __launch_bounds__(256) void max_out_kernel(
    const float* __restrict__ a, const float* __restrict__ b,
    float* __restrict__ out, int n) {
    int i = blockIdx.x * 256 + threadIdx.x;
    if (i < n) out[i] = fmaxf(a[i], b[i]);
}

// ---------------------------------------------------------------- launch
extern "C" void kernel_launch(void* const* d_in, const int* in_sizes, int n_in,
                              void* d_out, int out_size, void* d_ws, size_t ws_size,
                              hipStream_t stream) {
    (void)in_sizes; (void)n_in; (void)out_size; (void)ws_size;
    const float* R  = (const float*)d_in[0];
    const float* L  = (const float*)d_in[1];
    const float* Dg = (const float*)d_in[2];
    const float* De = (const float*)d_in[3];
    // transposed-weight source list: WQ_r, WK_r, WV_r, WQ_l, WK_l, WV_l, w1_r, w2_r, w1_l, w2_l
    const float* Wsrc[10] = {
        (const float*)d_in[4],  (const float*)d_in[5],  (const float*)d_in[6],
        (const float*)d_in[7],  (const float*)d_in[8],  (const float*)d_in[9],
        (const float*)d_in[10], (const float*)d_in[12], (const float*)d_in[14],
        (const float*)d_in[16]
    };

    char* p = (char*)d_ws;
    auto alloc = [&](size_t bytes) -> char* {
        char* r = p;
        p += (bytes + 255) & ~(size_t)255;
        return r;
    };
    const size_t ND_h = (size_t)NN * DD * sizeof(_Float16);
    const size_t ND_f = (size_t)NN * DD * sizeof(float);
    const size_t W_h  = (size_t)DD * DD * sizeof(_Float16);

    _Float16* Rh = (_Float16*)alloc(ND_h);
    _Float16* Lh = (_Float16*)alloc(ND_h);
    _Float16* Wt[10];
    for (int i = 0; i < 10; ++i) Wt[i] = (_Float16*)alloc(W_h);
    _Float16* Qh   = (_Float16*)alloc(ND_h);
    _Float16* Kh2  = (_Float16*)alloc(ND_h);
    _Float16* VTh  = (_Float16*)alloc(ND_h);
    float*    attn = (float*)alloc(ND_f);
    float*    y1f  = (float*)alloc(ND_f);
    _Float16* y1h  = (_Float16*)alloc(ND_h);
    _Float16* hh   = (_Float16*)alloc(ND_h);
    float*    f2   = (float*)alloc(ND_f);
    float*    res0 = (float*)alloc(ND_f);
    float*    res1 = (float*)alloc(ND_f);

    const int nElem = NN * DD;
    f32_to_f16_kernel<<<nElem / 256, 256, 0, stream>>>(R, Rh, nElem);
    f32_to_f16_kernel<<<nElem / 256, 256, 0, stream>>>(L, Lh, nElem);
    for (int i = 0; i < 10; ++i)
        transpose_w_kernel<<<dim3(DD / 32, DD / 32), dim3(32, 32), 0, stream>>>(Wsrc[i], Wt[i], DD);

    dim3 gemmGrid(DD / 64, NN / 64);
    for (int br = 0; br < 2; ++br) {
        const _Float16* xq_h  = (br == 0) ? Lh : Rh;
        const _Float16* xkv_h = (br == 0) ? Rh : Lh;
        const float*    xkv   = (br == 0) ? R  : L;
        const _Float16* WQt = Wt[br == 0 ? 0 : 3];
        const _Float16* WKt = Wt[br == 0 ? 1 : 4];
        const _Float16* WVt = Wt[br == 0 ? 2 : 5];
        const _Float16* W1t = Wt[br == 0 ? 6 : 8];
        const _Float16* W2t = Wt[br == 0 ? 7 : 9];
        const float* b1 = (const float*)d_in[br == 0 ? 11 : 15];
        const float* b2 = (const float*)d_in[br == 0 ? 13 : 17];
        const float* g1 = (const float*)d_in[br == 0 ? 18 : 22];
        const float* e1 = (const float*)d_in[br == 0 ? 19 : 23];
        const float* g2 = (const float*)d_in[br == 0 ? 20 : 24];
        const float* e2 = (const float*)d_in[br == 0 ? 21 : 25];
        const float* biasMat = (br == 0) ? De : Dg;
        float* res = (br == 0) ? res0 : res1;

        gemm_wmma_kernel<<<gemmGrid, 128, 0, stream>>>(
            xq_h, WQt, nullptr, Qh, nullptr, nullptr, NN, DD, DD, 0);
        gemm_wmma_kernel<<<gemmGrid, 128, 0, stream>>>(
            xkv_h, WKt, nullptr, Kh2, nullptr, nullptr, NN, DD, DD, 0);
        gemm_wmma_kernel<<<gemmGrid, 128, 0, stream>>>(
            xkv_h, WVt, nullptr, nullptr, VTh, nullptr, NN, DD, DD, 0);

        flash_attn_kernel<<<NN / 16, 128, 0, stream>>>(
            Qh, Kh2, VTh, biasMat, attn, (br == 0) ? 1 : 0);

        resid_ln_kernel<<<NN, 256, 0, stream>>>(attn, xkv, g1, e1, y1f, y1h);

        gemm_wmma_kernel<<<gemmGrid, 128, 0, stream>>>(
            y1h, W1t, b1, hh, nullptr, nullptr, NN, DD, DD, 1);
        gemm_wmma_kernel<<<gemmGrid, 128, 0, stream>>>(
            hh, W2t, b2, nullptr, nullptr, f2, NN, DD, DD, 0);

        resid_ln_kernel<<<NN, 256, 0, stream>>>(y1f, f2, g2, e2, res, nullptr);
    }

    max_out_kernel<<<nElem / 256, 256, 0, stream>>>(res0, res1, (float*)d_out, nElem);
}
